// CommonModel_50826642981332
// MI455X (gfx1250) — matrix operand order repaired and checked by reading, not compiled, not run on hardware
//
#include <hip/hip_runtime.h>

typedef unsigned short ushort_t;
typedef __attribute__((ext_vector_type(16))) __bf16 v16bf;
typedef __attribute__((ext_vector_type(8)))  float  v8f;

union FragBF { v16bf v; uint4 q[2]; };

__device__ inline ushort_t f2bf(float f) {
    unsigned u = __float_as_uint(f);
    unsigned r = (u + 0x7FFFu + ((u >> 16) & 1u)) >> 16;   // RNE
    return (ushort_t)r;
}
__device__ inline unsigned ordenc(float f) {
    unsigned u = __float_as_uint(f);
    return (u & 0x80000000u) ? ~u : (u | 0x80000000u);
}
__device__ inline float orddec(unsigned k) {
    unsigned u = (k & 0x80000000u) ? (k & 0x7FFFFFFFu) : ~k;
    return __uint_as_float(u);
}

// ---------------------------------------------------------------------------
// Pack a 128x128 f32 weight matrix into bf16 WMMA B-fragment order:
// for k-chunk c (0..3), n-tile t (0..7), lane l (0..31):
//   16 contiguous bf16 = W[kb+j, col], kb = c*32 + (l/16)*16, col = t*16 + l%16
// ---------------------------------------------------------------------------
__global__ void k_pack_w(const float* __restrict__ W, ushort_t* __restrict__ P) {
    int tid = blockIdx.x * blockDim.x + threadIdx.x;
    if (tid >= 1024) return;
    int l = tid & 31, t = (tid >> 5) & 7, c = tid >> 8;
    int col = t * 16 + (l & 15);
    int kb  = c * 32 + (l >> 4) * 16;
    ushort_t* dst = P + ((size_t)((c * 8 + t) * 32 + l)) * 16;
    #pragma unroll
    for (int j = 0; j < 16; ++j) dst[j] = f2bf(W[(kb + j) * 128 + col]);
}

// ---------------------------------------------------------------------------
// Y[N,128] = act( A@W1 (+ A2@W2) + bias ), bf16 operands / f32 accumulate.
// Block = 128 threads = 4 waves; each wave -> 16 rows x 128 cols.
// ---------------------------------------------------------------------------
__global__ void __launch_bounds__(128)
k_gemm(const ushort_t* __restrict__ A,  const ushort_t* __restrict__ Wpk,
       const ushort_t* __restrict__ A2, const ushort_t* __restrict__ Wpk2,
       const float* __restrict__ bias, float* __restrict__ Y, int N, int relu) {
    int wave = threadIdx.x >> 5;
    int lane = threadIdx.x & 31;
    int row0 = blockIdx.x * 64 + wave * 16;
    int lhalf = lane >> 4;
    int lmod  = lane & 15;

    v8f acc[8];
    #pragma unroll
    for (int t = 0; t < 8; ++t)
        #pragma unroll
        for (int r = 0; r < 8; ++r) acc[t][r] = 0.0f;

    int rowA = row0 + lmod;
    for (int pass = 0; pass < 2; ++pass) {
        const ushort_t* Ap = pass ? A2 : A;
        const ushort_t* Wp = pass ? Wpk2 : Wpk;
        if (Ap == nullptr) continue;            // uniform branch
        #pragma unroll
        for (int kc = 0; kc < 4; ++kc) {
            FragBF a;
            if (rowA < N) {
                const uint4* ap = (const uint4*)(Ap + (size_t)rowA * 128 + kc * 32 + lhalf * 16);
                a.q[0] = ap[0];                 // K = kc*32 + base .. +7
                a.q[1] = ap[1];                 // K = kc*32 + base + 8 .. +15
            } else {
                a.q[0] = make_uint4(0, 0, 0, 0);
                a.q[1] = make_uint4(0, 0, 0, 0);
            }
            #pragma unroll
            for (int t = 0; t < 8; ++t) {
                FragBF b;
                const uint4* bp = (const uint4*)(Wp + (size_t)((kc * 8 + t) * 32 + lane) * 16);
                b.q[0] = bp[0];
                b.q[1] = bp[1];
                acc[t] = __builtin_amdgcn_wmma_f32_16x16x32_bf16(
                    false, a.v, false, b.v, (short)0, acc[t], false, false);
            }
        }
    }
    #pragma unroll
    for (int t = 0; t < 8; ++t) {
        int col = t * 16 + lmod;
        float bv = bias ? bias[col] : 0.0f;
        #pragma unroll
        for (int r = 0; r < 8; ++r) {
            int row = row0 + lhalf * 8 + r;
            if (row < N) {
                float v = acc[t][r] + bv;
                if (relu) v = v > 0.0f ? v : 0.0f;
                Y[(size_t)row * 128 + col] = v;
            }
        }
    }
}

// ------------------------- small elementwise/edge kernels -------------------
__global__ void k_count(const int* __restrict__ dst, float* __restrict__ c, int E) {
    int e = blockIdx.x * blockDim.x + threadIdx.x;
    if (e < E) atomicAdd(&c[dst[e]], 1.0f);
}
__global__ void k_dinv(const float* __restrict__ deg, float* __restrict__ dinv, int N) {
    int n = blockIdx.x * blockDim.x + threadIdx.x;
    if (n < N) dinv[n] = deg[n] > 0.0f ? rsqrtf(fmaxf(deg[n], 1.0f)) : 0.0f;
}
__global__ void k_tag_hop(const int* __restrict__ src, const int* __restrict__ dst,
                          const float* __restrict__ dinv, const float* __restrict__ h,
                          float* __restrict__ hn, int E, int D) {
    int e = blockIdx.x * blockDim.x + threadIdx.x;
    if (e >= E) return;
    int s = src[e], d = dst[e];
    float nrm = dinv[s] * dinv[d];
    for (int i = 0; i < D; ++i) atomicAdd(&hn[(size_t)d * D + i], h[(size_t)s * D + i] * nrm);
}
__global__ void k_tag_accum(const float* __restrict__ h, const float* __restrict__ W,
                            const float* __restrict__ bias, float* __restrict__ out,
                            int N, int D, int first, int last) {
    int tid = blockIdx.x * blockDim.x + threadIdx.x;
    if (tid >= N * 128) return;
    int n = tid >> 7, c = tid & 127;
    float acc = first ? 0.0f : out[tid];
    for (int i = 0; i < D; ++i) acc += h[(size_t)n * D + i] * W[i * 128 + c];
    if (last) { acc += bias[c]; acc = acc > 0.0f ? acc : 0.0f; }
    out[tid] = acc;
}
__global__ void k_scatter_feat(const int* __restrict__ src, const int* __restrict__ dst,
                               const float* __restrict__ x, float* __restrict__ s, int E) {
    int tid = blockIdx.x * blockDim.x + threadIdx.x;
    int e = tid >> 5;
    if (e >= E) return;
    int g = tid & 31;
    int sv = src[e], dv = dst[e];
    float4 v = ((const float4*)(x + (size_t)sv * 128))[g];
    float* o = s + (size_t)dv * 128 + g * 4;
    atomicAdd(o + 0, v.x); atomicAdd(o + 1, v.y);
    atomicAdd(o + 2, v.z); atomicAdd(o + 3, v.w);
}
__global__ void k_mean_bf16(const float* __restrict__ s, const float* __restrict__ cnt,
                            ushort_t* __restrict__ y, int total) {
    int tid = blockIdx.x * blockDim.x + threadIdx.x;
    if (tid >= total) return;
    int n = tid >> 7;
    y[tid] = f2bf(s[tid] / fmaxf(cnt[n], 1.0f));
}
__global__ void k_f32_to_bf16(const float* __restrict__ x, ushort_t* __restrict__ y, int n) {
    int tid = blockIdx.x * blockDim.x + threadIdx.x;
    if (tid < n) y[tid] = f2bf(x[tid]);
}
__global__ void k_rowdot(const float* __restrict__ X, const float* __restrict__ w,
                         const float* __restrict__ bptr, float* __restrict__ out, int N) {
    int n = blockIdx.x * blockDim.x + threadIdx.x;
    if (n >= N) return;
    const float4* x4 = (const float4*)(X + (size_t)n * 128);
    const float4* w4 = (const float4*)w;
    float acc = bptr ? bptr[0] : 0.0f;
    #pragma unroll
    for (int i = 0; i < 32; ++i) {
        float4 a = x4[i], b = w4[i];
        acc += a.x * b.x + a.y * b.y + a.z * b.z + a.w * b.w;
    }
    out[n] = acc;
}
__global__ void k_dot128(const float* __restrict__ a, const float* __restrict__ b,
                         float* __restrict__ out) {
    __shared__ float sm[128];
    int t = threadIdx.x;
    sm[t] = a[t] * b[t];
    __syncthreads();
    for (int s = 64; s > 0; s >>= 1) { if (t < s) sm[t] += sm[t + s]; __syncthreads(); }
    if (t == 0) out[0] = sm[0];
}
__global__ void k_edge_score(const int* __restrict__ src, const int* __restrict__ dst,
                             const float* __restrict__ ea, const float* __restrict__ an_s,
                             const float* __restrict__ an_d, const float* __restrict__ cscal,
                             float* __restrict__ a, unsigned* __restrict__ m_u, int E) {
    int e = blockIdx.x * blockDim.x + threadIdx.x;
    if (e >= E) return;
    int s = src[e], d = dst[e];
    float v = an_s[s] + an_d[d] + ea[e] * cscal[0];
    v = v > 0.0f ? v : 0.2f * v;                 // leaky_relu 0.2
    a[e] = v;
    atomicMax(&m_u[d], ordenc(v));
}
__global__ void k_edge_exp(const int* __restrict__ dst, float* __restrict__ a,
                           const unsigned* __restrict__ m_u, float* __restrict__ denom, int E) {
    int e = blockIdx.x * blockDim.x + threadIdx.x;
    if (e >= E) return;
    int d = dst[e];
    float v = __expf(a[e] - orddec(m_u[d]));
    a[e] = v;
    atomicAdd(&denom[d], v);
}
__global__ void k_alpha_scatter(const int* __restrict__ src, const int* __restrict__ dst,
                                const float* __restrict__ ev, const float* __restrict__ denom,
                                const float* __restrict__ hs, float* __restrict__ out, int E) {
    int tid = blockIdx.x * blockDim.x + threadIdx.x;
    int e = tid >> 5;
    if (e >= E) return;
    int g = tid & 31;
    int s = src[e], d = dst[e];
    float alpha = ev[e] / fmaxf(denom[d], 1e-16f);
    float4 v = ((const float4*)(hs + (size_t)s * 128))[g];
    float* o = out + (size_t)d * 128 + g * 4;
    atomicAdd(o + 0, alpha * v.x); atomicAdd(o + 1, alpha * v.y);
    atomicAdd(o + 2, alpha * v.z); atomicAdd(o + 3, alpha * v.w);
}
__global__ void k_bias_relu(float* __restrict__ x, const float* __restrict__ b, int total) {
    int tid = blockIdx.x * blockDim.x + threadIdx.x;
    if (tid >= total) return;
    float v = x[tid] + b[tid & 127];
    x[tid] = v > 0.0f ? v : 0.0f;
}

// ---------------------------------------------------------------------------
extern "C" void kernel_launch(void* const* d_in, const int* in_sizes, int n_in,
                              void* d_out, int out_size, void* d_ws, size_t ws_size,
                              hipStream_t stream) {
    (void)n_in; (void)out_size; (void)ws_size;
    const int H   = 128;
    const int NV  = in_sizes[0] / 5;
    const int NS  = in_sizes[1] / 6;
    const int EVV = in_sizes[2] / 2;
    const int EH  = in_sizes[4] / 2;
    const int EIN = in_sizes[6] / 2;
    const int ESS = in_sizes[7] / 2;

    const float* game_x  = (const float*)d_in[0];
    const float* state_x = (const float*)d_in[1];
    const int*   evv     = (const int*)d_in[2];
    const int*   ehist   = (const int*)d_in[4];
    const float* ea_hist = (const float*)d_in[5];
    const int*   ein     = (const int*)d_in[6];
    const int*   ess     = (const int*)d_in[7];
    const float* tag1W   = (const float*)d_in[8];
    const float* tag1b   = (const float*)d_in[9];
    const float* tag2W   = (const float*)d_in[10];
    const float* tag2b   = (const float*)d_in[11];
    const float* sageWl  = (const float*)d_in[12];
    const float* sagebl  = (const float*)d_in[13];
    const float* sageWr  = (const float*)d_in[14];
    const float* gWsrc   = (const float*)d_in[15];
    const float* gWdst   = (const float*)d_in[16];
    const float* gWe     = (const float*)d_in[17];
    const float* gasrc   = (const float*)d_in[18];
    const float* gadst   = (const float*)d_in[19];
    const float* gae     = (const float*)d_in[20];
    const float* gb      = (const float*)d_in[21];
    const float* mlpW    = (const float*)d_in[22];
    const float* mlpb    = (const float*)d_in[23];

    char* ws = (char*)d_ws;
    size_t off = 0;
    auto alloc = [&](size_t bytes) -> char* {
        char* p = ws + off;
        off = (off + bytes + 255) & ~(size_t)255;
        return p;
    };
    float*    BV_A   = (float*)alloc((size_t)NV * H * 4);
    float*    BV_B   = (float*)alloc((size_t)NV * H * 4);
    float*    BS_A   = (float*)alloc((size_t)NS * H * 4);
    float*    BS_B   = (float*)alloc((size_t)NS * H * 4);
    ushort_t* BF0    = (ushort_t*)alloc((size_t)NV * H * 2);
    ushort_t* BF1    = (ushort_t*)alloc((size_t)NV * H * 2);
    float*    deg_v  = (float*)alloc((size_t)NV * 4);
    float*    dinv_v = (float*)alloc((size_t)NV * 4);
    float*    deg_s  = (float*)alloc((size_t)NS * 4);
    float*    dinv_s = (float*)alloc((size_t)NS * 4);
    float*    cnt    = (float*)alloc((size_t)NV * 4);
    float*    hvA    = (float*)alloc((size_t)NV * 5 * 4);
    float*    hvB    = (float*)alloc((size_t)NV * 5 * 4);
    float*    hsA    = (float*)alloc((size_t)NS * 6 * 4);
    float*    hsB    = (float*)alloc((size_t)NS * 6 * 4);
    float*    an_src = (float*)alloc((size_t)NV * 4);
    float*    an_dst = (float*)alloc((size_t)NS * 4);
    unsigned* m_u    = (unsigned*)alloc((size_t)NS * 4);
    float*    denom  = (float*)alloc((size_t)NS * 4);
    float*    escore = (float*)alloc((size_t)EH * 4);
    float*    cscal  = (float*)alloc(256);
    ushort_t* PK     = (ushort_t*)alloc((size_t)16 * H * H * 2);

    auto cdiv = [](long long a, long long b) -> int { return (int)((a + b - 1) / b); };

    // ---- pack weights into WMMA bf16 fragment layout (L2-resident) ----
    for (int i = 0; i < 7; ++i) {
        k_pack_w<<<4, 256, 0, stream>>>(sageWl + (size_t)i * H * H, PK + (size_t)i * H * H);
        k_pack_w<<<4, 256, 0, stream>>>(sageWr + (size_t)i * H * H, PK + (size_t)(7 + i) * H * H);
    }
    k_pack_w<<<4, 256, 0, stream>>>(gWsrc, PK + (size_t)14 * H * H);
    k_pack_w<<<4, 256, 0, stream>>>(gWdst, PK + (size_t)15 * H * H);

    // ---- gcn_norm degrees ----
    hipMemsetAsync(deg_v, 0, (size_t)NV * 4, stream);
    k_count<<<cdiv(EVV, 256), 256, 0, stream>>>(evv + EVV, deg_v, EVV);
    k_dinv<<<cdiv(NV, 256), 256, 0, stream>>>(deg_v, dinv_v, NV);
    hipMemsetAsync(deg_s, 0, (size_t)NS * 4, stream);
    k_count<<<cdiv(ESS, 256), 256, 0, stream>>>(ess + ESS, deg_s, ESS);
    k_dinv<<<cdiv(NS, 256), 256, 0, stream>>>(deg_s, dinv_s, NS);

    // ---- TAGConv (game) -> BV_A ----
    {
        int gN = cdiv((long long)NV * H, 256), gE = cdiv(EVV, 256);
        k_tag_accum<<<gN, 256, 0, stream>>>(game_x, tag1W, nullptr, BV_A, NV, 5, 1, 0);
        hipMemsetAsync(hvA, 0, (size_t)NV * 5 * 4, stream);
        k_tag_hop<<<gE, 256, 0, stream>>>(evv, evv + EVV, dinv_v, game_x, hvA, EVV, 5);
        k_tag_accum<<<gN, 256, 0, stream>>>(hvA, tag1W + 1 * 5 * H, nullptr, BV_A, NV, 5, 0, 0);
        hipMemsetAsync(hvB, 0, (size_t)NV * 5 * 4, stream);
        k_tag_hop<<<gE, 256, 0, stream>>>(evv, evv + EVV, dinv_v, hvA, hvB, EVV, 5);
        k_tag_accum<<<gN, 256, 0, stream>>>(hvB, tag1W + 2 * 5 * H, nullptr, BV_A, NV, 5, 0, 0);
        hipMemsetAsync(hvA, 0, (size_t)NV * 5 * 4, stream);
        k_tag_hop<<<gE, 256, 0, stream>>>(evv, evv + EVV, dinv_v, hvB, hvA, EVV, 5);
        k_tag_accum<<<gN, 256, 0, stream>>>(hvA, tag1W + 3 * 5 * H, tag1b, BV_A, NV, 5, 0, 1);
    }
    // ---- TAGConv (state) -> BS_A ----
    {
        int gN = cdiv((long long)NS * H, 256), gE = cdiv(ESS, 256);
        k_tag_accum<<<gN, 256, 0, stream>>>(state_x, tag2W, nullptr, BS_A, NS, 6, 1, 0);
        hipMemsetAsync(hsA, 0, (size_t)NS * 6 * 4, stream);
        k_tag_hop<<<gE, 256, 0, stream>>>(ess, ess + ESS, dinv_s, state_x, hsA, ESS, 6);
        k_tag_accum<<<gN, 256, 0, stream>>>(hsA, tag2W + 1 * 6 * H, nullptr, BS_A, NS, 6, 0, 0);
        hipMemsetAsync(hsB, 0, (size_t)NS * 6 * 4, stream);
        k_tag_hop<<<gE, 256, 0, stream>>>(ess, ess + ESS, dinv_s, hsA, hsB, ESS, 6);
        k_tag_accum<<<gN, 256, 0, stream>>>(hsB, tag2W + 2 * 6 * H, nullptr, BS_A, NS, 6, 0, 0);
        hipMemsetAsync(hsA, 0, (size_t)NS * 6 * 4, stream);
        k_tag_hop<<<gE, 256, 0, stream>>>(ess, ess + ESS, dinv_s, hsB, hsA, ESS, 6);
        k_tag_accum<<<gN, 256, 0, stream>>>(hsA, tag2W + 3 * 6 * H, tag2b, BS_A, NS, 6, 0, 1);
    }

    // ---- SAGEConv layer (mean aggregate + dual WMMA GEMM) ----
    auto sage_layer = [&](const float* xsrc, const float* xdst, int n_dst,
                          const int* src, const int* dst, int E, int li, float* out) {
        hipMemsetAsync(out, 0, (size_t)n_dst * H * 4, stream);
        hipMemsetAsync(cnt, 0, (size_t)n_dst * 4, stream);
        k_scatter_feat<<<cdiv((long long)E * 32, 256), 256, 0, stream>>>(src, dst, xsrc, out, E);
        k_count<<<cdiv(E, 256), 256, 0, stream>>>(dst, cnt, E);
        k_mean_bf16<<<cdiv((long long)n_dst * H, 256), 256, 0, stream>>>(out, cnt, BF0, n_dst * H);
        k_f32_to_bf16<<<cdiv((long long)n_dst * H, 256), 256, 0, stream>>>(xdst, BF1, n_dst * H);
        k_gemm<<<cdiv(n_dst, 64), 128, 0, stream>>>(BF0, PK + (size_t)li * H * H,
                                                    BF1, PK + (size_t)(7 + li) * H * H,
                                                    sagebl + (size_t)li * H, out, n_dst, 1);
    };

    // game tower
    sage_layer(BV_A, BV_A, NV, evv, evv + EVV, EVV, 0, BV_B);
    sage_layer(BV_B, BV_B, NV, evv, evv + EVV, EVV, 1, BV_A);   // gx = BV_A
    // state tower
    sage_layer(BS_A, BS_A, NS, ess, ess + ESS, ESS, 2, BS_B);
    sage_layer(BS_B, BS_B, NS, ess, ess + ESS, ESS, 3, BS_A);   // sx = BS_A

    // ---- GATConv (game -> state) ----
    k_f32_to_bf16<<<cdiv((long long)NV * H, 256), 256, 0, stream>>>(BV_A, BF0, NV * H);
    k_gemm<<<cdiv(NV, 64), 128, 0, stream>>>(BF0, PK + (size_t)14 * H * H,
                                             nullptr, nullptr, nullptr, BV_B, NV, 0);  // hs
    k_rowdot<<<cdiv(NV, 256), 256, 0, stream>>>(BV_B, gasrc, nullptr, an_src, NV);
    k_f32_to_bf16<<<cdiv((long long)NS * H, 256), 256, 0, stream>>>(BS_A, BF0, NS * H);
    k_gemm<<<cdiv(NS, 64), 128, 0, stream>>>(BF0, PK + (size_t)15 * H * H,
                                             nullptr, nullptr, nullptr, BS_B, NS, 0);  // hd
    k_rowdot<<<cdiv(NS, 256), 256, 0, stream>>>(BS_B, gadst, nullptr, an_dst, NS);
    k_dot128<<<1, 128, 0, stream>>>(gWe, gae, cscal);
    hipMemsetAsync(m_u, 0, (size_t)NS * 4, stream);              // 0 == minimal ordered key
    k_edge_score<<<cdiv(EH, 256), 256, 0, stream>>>(ehist, ehist + EH, ea_hist,
                                                    an_src, an_dst, cscal, escore, m_u, EH);
    hipMemsetAsync(denom, 0, (size_t)NS * 4, stream);
    k_edge_exp<<<cdiv(EH, 256), 256, 0, stream>>>(ehist + EH, escore, m_u, denom, EH);
    hipMemsetAsync(BS_B, 0, (size_t)NS * H * 4, stream);         // hist accumulator (hd dead)
    k_alpha_scatter<<<cdiv((long long)EH * 32, 256), 256, 0, stream>>>(ehist, ehist + EH,
                                                                      escore, denom, BV_B, BS_B, EH);
    k_bias_relu<<<cdiv((long long)NS * H, 256), 256, 0, stream>>>(BS_B, gb, NS * H);  // hist = BS_B

    // in_x = relu(sage(gx, hist)) over "in" bipartite edges -> BS_A
    sage_layer(BV_A, BS_B, NS, ein, ein + EIN, EIN, 4, BS_A);
    // second state tower
    sage_layer(BS_A, BS_A, NS, ess, ess + ESS, ESS, 5, BS_B);
    sage_layer(BS_B, BS_B, NS, ess, ess + ESS, ESS, 6, BS_A);

    // ---- final MLP (H -> 1) ----
    k_rowdot<<<cdiv(NS, 256), 256, 0, stream>>>(BS_A, mlpW, mlpb, (float*)d_out, NS);
}
